// SimplifiedGNN_IDS_2748779070199
// MI455X (gfx1250) — matrix operand order repaired and checked by reading, not compiled, not run on hardware
//
#include <hip/hip_runtime.h>
#include <hip/hip_bf16.h>

typedef _Float16 h16 __attribute__((ext_vector_type(16)));
typedef _Float16 h8  __attribute__((ext_vector_type(8)));
typedef float    f8  __attribute__((ext_vector_type(8)));
typedef unsigned int u32x4 __attribute__((ext_vector_type(4)));
typedef unsigned int u32x8 __attribute__((ext_vector_type(8)));

#define B_GRAPHS 32768
#define N_NODES  16
#define ND       64
#define HID      128
#define NHEAD    4
#define DHEAD    32
#define NC       10
#define WPB      3        // waves (graphs) per block -> 57.6KB static LDS
#define HSTR     136      // padded row stride (halfs) for [16][128] f16 bufs
#define XSTR     72       // padded row stride (halfs) for [16][64] x staging / attn buf
#define VSTR     24       // padded row stride (halfs) for vT [128 cols][16 nodes]

// f16 weight workspace layout (offsets in halfs)
#define OFF_ENC  0
#define OFF_Q    (128*64)
#define OFF_K    (OFF_Q + 2*128*128)
#define OFF_V    (OFF_K + 2*128*128)
#define OFF_W1   (OFF_V + 2*128*128)
#define OFF_W2   (OFF_W1 + 128*256)
#define WS_HALFS (OFF_W2 + 16*128)

__device__ __forceinline__ f8 wmma16(h16 a, h16 b, f8 c) {
  // D = A(16x32 f16) * B(32x16 f16) + C(16x16 f32)
  return __builtin_amdgcn_wmma_f32_16x16x32_f16(false, a, false, b, (short)0, c,
                                                false, false);
}

// ---- Tensor Data Mover: 1-D contiguous copy global -> LDS (per-wave, TENSORcnt)
// D# group0: count=1 | lds_addr[63:32] | global_addr[120:64] | type[127:126]=2
// D# group1: data_size[17:16] | tensor_dim0[79:48] | tensor_dim1[111:80]
//            | tile_dim0[127:112] | tensor_dim0_stride[207:160]
__device__ __forceinline__ void tdm_load_1d_f32(void* lds_dst, const void* gsrc,
                                                unsigned elems) {
  unsigned lds_addr = (unsigned)(unsigned long long)lds_dst;  // low 32b = LDS offset
  unsigned long long ga = (unsigned long long)gsrc;
  u32x4 g0;
  g0[0] = 1u;                                   // count=1, user mode
  g0[1] = lds_addr;                             // lds_addr (bytes)
  g0[2] = (unsigned)(ga & 0xffffffffu);         // global_addr[31:0]
  g0[3] = (unsigned)((ga >> 32) & 0x01ffffffu)  // global_addr[56:32]
          | (2u << 30);                         // type = 2 ("image")
  u32x8 g1;
  g1[0] = (2u << 16);                           // data_size = 2 -> 4 bytes
  g1[1] = (elems & 0xffffu) << 16;              // tensor_dim0[15:0]
  g1[2] = ((elems >> 16) & 0xffffu)             // tensor_dim0[31:16]
          | (1u << 16);                         // tensor_dim1 = 1 (low half)
  g1[3] = (elems & 0xffffu) << 16;              // tile_dim0 (<= 65535)
  g1[4] = 0u;                                   // tile_dim1 = 0, tile_dim2 = 0
  g1[5] = elems;                                // tensor_dim0_stride[31:0]
  g1[6] = 0u;                                   // stride[47:32] | dim1_stride lo
  g1[7] = 0u;
  asm volatile("tensor_load_to_lds %0, %1" ::"s"(g0), "s"(g1) : "memory");
}

// A fragment (16x32, MxK) from row-major f16 LDS buffer, row stride S halfs.
// Lane l: m = l&15; K(l,j) = (l>>4)*16 + j  (contiguous, matches B)
__device__ __forceinline__ h16 load_a_lds(const _Float16* base, int S, int kbase, int lane) {
  const int m = lane & 15, hi = lane >> 4;
  const _Float16* p = base + m * S + kbase + hi * 16;
  h8 lo = *(const h8*)p;
  h8 up = *(const h8*)(p + 8);
  h16 a;
#pragma unroll
  for (int i = 0; i < 8; i++) { a[i] = lo[i]; a[8 + i] = up[i]; }
  return a;
}

// B fragment (32x16, KxN) from global f16 weights pre-transposed to [N][K].
// Lane l: n = l&15; K(l,j) = (l>>4)*16 + j  -> contiguous 32B per lane.
__device__ __forceinline__ h16 load_b_gl(const _Float16* Wt, int K, int ntile, int kbase,
                                         int lane) {
  const int n = (ntile << 4) + (lane & 15), hi = lane >> 4;
  const _Float16* p = Wt + n * K + kbase + hi * 16;
  h8 lo = *(const h8*)p;
  h8 up = *(const h8*)(p + 8);
  h16 b;
#pragma unroll
  for (int i = 0; i < 8; i++) { b[i] = lo[i]; b[8 + i] = up[i]; }
  return b;
}

// C fragment (lane n = l&15, row m = (l>>4)*8 + r) -> row-major f16 LDS
__device__ __forceinline__ void store_c_rowmajor(f8 c, _Float16* base, int S, int ntile,
                                                 int lane) {
  const int n = (ntile << 4) + (lane & 15), hi = lane >> 4;
#pragma unroll
  for (int r = 0; r < 8; r++) base[(hi * 8 + r) * S + n] = (_Float16)c[r];
}

// C fragment -> vT[col][node] (contiguous b128 store per lane)
__device__ __forceinline__ void store_c_colmajor(f8 c, _Float16* vT, int ntile, int lane) {
  const int col = (ntile << 4) + (lane & 15), hi = lane >> 4;
  h8 p;
#pragma unroll
  for (int r = 0; r < 8; r++) p[r] = (_Float16)c[r];
  *(h8*)(vT + col * VSTR + hi * 8) = p;
}

// ---------------- weight prep: fp32 -> f16, pre-transposed --------------------
__global__ void prep_weights_kernel(const float* __restrict__ encW,
                                    const float* __restrict__ Wq,
                                    const float* __restrict__ Wk,
                                    const float* __restrict__ Wv,
                                    const float* __restrict__ W1,
                                    const float* __restrict__ W2,
                                    _Float16* __restrict__ ws) {
  for (int i = blockIdx.x * blockDim.x + threadIdx.x; i < WS_HALFS;
       i += gridDim.x * blockDim.x) {
    float v;
    if (i < OFF_Q) {                       // encT[n=128][k=64]
      int n = i >> 6, k = i & 63;
      v = encW[k * HID + n];
    } else if (i < OFF_K) {                // qT[L][n=128][k=128]
      int t = i - OFF_Q; int L = t >> 14; t &= 16383; int n = t >> 7, k = t & 127;
      v = Wq[L * HID * HID + k * HID + n];
    } else if (i < OFF_V) {
      int t = i - OFF_K; int L = t >> 14; t &= 16383; int n = t >> 7, k = t & 127;
      v = Wk[L * HID * HID + k * HID + n];
    } else if (i < OFF_W1) {
      int t = i - OFF_V; int L = t >> 14; t &= 16383; int n = t >> 7, k = t & 127;
      v = Wv[L * HID * HID + k * HID + n];
    } else if (i < OFF_W2) {               // w1T[j=128][k=256]
      int t = i - OFF_W1; int j = t >> 8, k = t & 255;
      v = W1[k * HID + j];
    } else {                               // w2T[c=16][k=128], pad c>=10 with 0
      int t = i - OFF_W2; int c = t >> 7, k = t & 127;
      v = (c < NC) ? W2[k * NC + c] : 0.0f;
    }
    ws[i] = (_Float16)v;
  }
}

// ---------------- main kernel: one wave == one graph --------------------------
__global__ __launch_bounds__(WPB * 32) void gnn_kernel(
    const float* __restrict__ x, const float* __restrict__ enc_b,
    const float* __restrict__ bq, const float* __restrict__ bk,
    const float* __restrict__ bv, const float* __restrict__ ln_g,
    const float* __restrict__ ln_b, const float* __restrict__ b1,
    const float* __restrict__ b2, const _Float16* __restrict__ ws,
    float* __restrict__ out) {
  __shared__ __align__(16) _Float16 s_hH[WPB][16 * HSTR];   // h f16 (A src); attn reuse
  __shared__ __align__(16) _Float16 s_qH[WPB][16 * HSTR];   // x f16 / Q / hid reuse
  __shared__ __align__(16) _Float16 s_kH[WPB][16 * HSTR];   // x TDM dst / K / pooled
  __shared__ __align__(16) _Float16 s_vT[WPB][128 * VSTR];  // V col-major

  const int wave = threadIdx.x >> 5;
  const int lane = threadIdx.x & 31;
  const int g = blockIdx.x * WPB + wave;
  if (g >= B_GRAPHS) return;              // wave-uniform (EXEC all-ones for WMMA)

  _Float16* hH = s_hH[wave];
  _Float16* qH = s_qH[wave];
  _Float16* kH = s_kH[wave];
  _Float16* vT = s_vT[wave];
  const int m = lane & 15, hi = lane >> 4;

  // ---- stage x[g] (16x64 fp32, one contiguous 4KB tile) via the Tensor Data
  //      Mover into the (currently idle) kH region, then convert to f16 [16][XSTR]
  tdm_load_1d_f32(kH, x + (size_t)g * (N_NODES * ND), N_NODES * ND);
  __builtin_amdgcn_s_wait_tensorcnt(0);
  {
    const float4* xf = (const float4*)kH;
#pragma unroll
    for (int t = lane; t < 256; t += 32) {
      float4 v = xf[t];
      int node = t >> 4;            // 16 float4 per 64-wide row
      int c = (t & 15) << 2;
      _Float16* p = qH + node * XSTR + c;
      p[0] = (_Float16)v.x; p[1] = (_Float16)v.y;
      p[2] = (_Float16)v.z; p[3] = (_Float16)v.w;
    }
  }

  // ---- encoder: h = x @ encW + enc_b, h kept as 8 register C-fragments
  f8 hfr[8];
#pragma unroll
  for (int t = 0; t < 8; t++) {
    float bias = enc_b[(t << 4) + m];
    f8 acc;
#pragma unroll
    for (int r = 0; r < 8; r++) acc[r] = bias;
#pragma unroll
    for (int ks = 0; ks < 2; ks++) {
      h16 a = load_a_lds(qH, XSTR, ks * 32, lane);
      h16 b = load_b_gl(ws + OFF_ENC, 64, t, ks * 32, lane);
      acc = wmma16(a, b, acc);
    }
    hfr[t] = acc;
  }

  const float scale = 0.17677669529663687f;  // 1/sqrt(32)

#pragma unroll
  for (int L = 0; L < 2; L++) {
    const _Float16* WqT = ws + OFF_Q + L * HID * HID;
    const _Float16* WkT = ws + OFF_K + L * HID * HID;
    const _Float16* WvT = ws + OFF_V + L * HID * HID;

    // warm GL2/L0 for the 3x32KB weight stream of this layer (1KB per lane)
    __builtin_prefetch(WqT + lane * 512, 0, 1);
    __builtin_prefetch(WkT + lane * 512, 0, 1);
    __builtin_prefetch(WvT + lane * 512, 0, 1);

    // h f16 into LDS for A-fragments
#pragma unroll
    for (int t = 0; t < 8; t++) store_c_rowmajor(hfr[t], hH, HSTR, t, lane);

    // hoist shared A-fragments (reused by Q, K, V across all 8 column tiles)
    h16 afr[4];
#pragma unroll
    for (int ks = 0; ks < 4; ks++) afr[ks] = load_a_lds(hH, HSTR, ks * 32, lane);

#pragma unroll
    for (int t = 0; t < 8; t++) {
      const int n = (t << 4) + m;
      f8 q, k, v;
      float bqv = bq[L * HID + n], bkv = bk[L * HID + n], bvv = bv[L * HID + n];
#pragma unroll
      for (int r = 0; r < 8; r++) { q[r] = bqv; k[r] = bkv; v[r] = bvv; }
#pragma unroll
      for (int ks = 0; ks < 4; ks++) {
        q = wmma16(afr[ks], load_b_gl(WqT, HID, t, ks * 32, lane), q);
        k = wmma16(afr[ks], load_b_gl(WkT, HID, t, ks * 32, lane), k);
        v = wmma16(afr[ks], load_b_gl(WvT, HID, t, ks * 32, lane), v);
      }
      store_c_rowmajor(q, qH, HSTR, t, lane);
      store_c_rowmajor(k, kH, HSTR, t, lane);
      store_c_colmajor(v, vT, t, lane);
    }

    // ---- attention: scores + softmax per head, attn stored f16 in hH region
    _Float16* sH = hH;  // [16 queries][XSTR]; col = hh*16 + key
#pragma unroll
    for (int hh = 0; hh < NHEAD; hh++) {
      h16 aq = load_a_lds(qH, HSTR, hh * DHEAD, lane);
      // B = K^T fragment: lane n = key node, K dim = d (contiguous in kH row)
      h16 bkf;
      {
        const _Float16* p = kH + m * HSTR + hh * DHEAD + hi * 16;
        h8 lo = *(const h8*)p;
        h8 up = *(const h8*)(p + 8);
#pragma unroll
        for (int i = 0; i < 8; i++) { bkf[i] = lo[i]; bkf[8 + i] = up[i]; }
      }
      f8 z = {};
      f8 s = wmma16(aq, bkf, z);
      // softmax across 16 keys (= across lanes within 16-lane half, per VGPR row)
#pragma unroll
      for (int r = 0; r < 8; r++) {
        float sv = s[r] * scale;
        float mx = sv;
        mx = fmaxf(mx, __shfl_xor(mx, 1));
        mx = fmaxf(mx, __shfl_xor(mx, 2));
        mx = fmaxf(mx, __shfl_xor(mx, 4));
        mx = fmaxf(mx, __shfl_xor(mx, 8));
        float e = __expf(sv - mx);
        float sum = e;
        sum += __shfl_xor(sum, 1);
        sum += __shfl_xor(sum, 2);
        sum += __shfl_xor(sum, 4);
        sum += __shfl_xor(sum, 8);
        float a = e / sum;
        sH[(hi * 8 + r) * XSTR + hh * 16 + m] = (_Float16)a;
      }
    }

    // ---- out = attn @ V + residual (residual rides in the C operand)
#pragma unroll
    for (int t = 0; t < 8; t++) {
      const int hh = t >> 1;
      // A = [attn(16x16) | 0]: lane group0 holds keys 0..15, group1 zero
      h16 a;
#pragma unroll
      for (int i = 0; i < 16; i++) a[i] = (_Float16)0.f;
      if (hi == 0) {
        h8 lo = *(const h8*)(sH + m * XSTR + hh * 16);
        h8 up = *(const h8*)(sH + m * XSTR + hh * 16 + 8);
#pragma unroll
        for (int i = 0; i < 8; i++) { a[i] = lo[i]; a[8 + i] = up[i]; }
      }
      // B = [V; 0]: K = node (group0 reads vT row, group1 zero)
      h16 b;
#pragma unroll
      for (int i = 0; i < 16; i++) b[i] = (_Float16)0.f;
      if (hi == 0) {
        const _Float16* p = vT + ((t << 4) + m) * VSTR;
        h8 lo = *(const h8*)p;
        h8 up = *(const h8*)(p + 8);
#pragma unroll
        for (int i = 0; i < 8; i++) { b[i] = lo[i]; b[8 + i] = up[i]; }
      }
      hfr[t] = wmma16(a, b, hfr[t]);   // out + h
    }

    // ---- layernorm over 128 cols per row, all in registers + shfl
    float gv[8], bvl[8];
#pragma unroll
    for (int t = 0; t < 8; t++) {
      int n = (t << 4) + m;
      gv[t] = ln_g[L * HID + n];
      bvl[t] = ln_b[L * HID + n];
    }
#pragma unroll
    for (int r = 0; r < 8; r++) {
      float s = 0.f;
#pragma unroll
      for (int t = 0; t < 8; t++) s += hfr[t][r];
      s += __shfl_xor(s, 1); s += __shfl_xor(s, 2);
      s += __shfl_xor(s, 4); s += __shfl_xor(s, 8);
      float mu = s * 0.0078125f;
      float vv = 0.f;
#pragma unroll
      for (int t = 0; t < 8; t++) { float d = hfr[t][r] - mu; vv += d * d; }
      vv += __shfl_xor(vv, 1); vv += __shfl_xor(vv, 2);
      vv += __shfl_xor(vv, 4); vv += __shfl_xor(vv, 8);
      float rs = rsqrtf(vv * 0.0078125f + 1e-5f);
#pragma unroll
      for (int t = 0; t < 8; t++)
        hfr[t][r] = (hfr[t][r] - mu) * rs * gv[t] + bvl[t];
    }
  }

  // ---- pooling: mean & max over the 16 nodes, per column (lanes hold columns)
  float* pooled = (float*)kH;  // 256 f32, kH region free now
#pragma unroll
  for (int t = 0; t < 8; t++) {
    float sm = 0.f, mxv = -1e30f;
#pragma unroll
    for (int r = 0; r < 8; r++) { float v = hfr[t][r]; sm += v; mxv = fmaxf(mxv, v); }
    sm += __shfl_xor(sm, 16);
    mxv = fmaxf(mxv, __shfl_xor(mxv, 16));
    if (hi == 0) {
      int n = (t << 4) + m;
      pooled[n] = sm * 0.0625f;
      pooled[HID + n] = mxv;
    }
  }

  // ---- MLP head (tiny: VALU f32 with f16 weights), hid = relu(pooled@W1 + b1)
  float* hidL = (float*)qH;  // 128 f32
  const _Float16* w1T = ws + OFF_W1;
#pragma unroll
  for (int jj = 0; jj < 4; jj++) {
    int j = lane + (jj << 5);
    float acc = b1[j];
    const _Float16* wp = w1T + j * 256;
#pragma unroll 4
    for (int k = 0; k < 256; k += 8) {
      h8 w = *(const h8*)(wp + k);
#pragma unroll
      for (int u = 0; u < 8; u++) acc += pooled[k + u] * (float)w[u];
    }
    hidL[j] = fmaxf(acc, 0.f);
  }

  if (lane < NC) {
    const _Float16* w2 = ws + OFF_W2 + lane * HID;
    float acc = b2[lane];
#pragma unroll 4
    for (int k = 0; k < HID; k += 8) {
      h8 w = *(const h8*)(w2 + k);
#pragma unroll
      for (int u = 0; u < 8; u++) acc += hidL[k + u] * (float)w[u];
    }
    out[(size_t)g * NC + lane] = acc;
  }
}

extern "C" void kernel_launch(void* const* d_in, const int* in_sizes, int n_in,
                              void* d_out, int out_size, void* d_ws, size_t ws_size,
                              hipStream_t stream) {
  (void)in_sizes; (void)n_in; (void)out_size; (void)ws_size;
  const float* x    = (const float*)d_in[0];
  const float* encW = (const float*)d_in[1];
  const float* encb = (const float*)d_in[2];
  const float* Wq   = (const float*)d_in[3];
  const float* bq   = (const float*)d_in[4];
  const float* Wk   = (const float*)d_in[5];
  const float* bk   = (const float*)d_in[6];
  const float* Wv   = (const float*)d_in[7];
  const float* bv   = (const float*)d_in[8];
  const float* lng  = (const float*)d_in[9];
  const float* lnb  = (const float*)d_in[10];
  const float* W1   = (const float*)d_in[11];
  const float* b1   = (const float*)d_in[12];
  const float* W2   = (const float*)d_in[13];
  const float* b2   = (const float*)d_in[14];
  _Float16* ws = (_Float16*)d_ws;
  float* out = (float*)d_out;

  prep_weights_kernel<<<128, 256, 0, stream>>>(encW, Wq, Wk, Wv, W1, W2, ws);

  const int grid = (B_GRAPHS + WPB - 1) / WPB;
  gnn_kernel<<<grid, WPB * 32, 0, stream>>>(x, encb, bq, bk, bv, lng, lnb, b1, b2,
                                            ws, out);
}
